// EncoderDecoder_59296318488642
// MI455X (gfx1250) — matrix operand order repaired and checked by reading, not compiled, not run on hardware
//
#include <hip/hip_runtime.h>
#include <hip/hip_bf16.h>

typedef __attribute__((ext_vector_type(16))) _Float16 v16h;
typedef __attribute__((ext_vector_type(8)))  float    v8f;

static constexpr int kH   = 32;   // hidden dim
static constexpr int kT   = 512;  // encoder sequence length
static constexpr int kDin = 4;    // input feature dim

// D = A(16x32 f16) x B(32x16 f16) + C(16x16 f32)
static __device__ __forceinline__ v8f wmma16(v16h a, v16h b, v8f c) {
  return __builtin_amdgcn_wmma_f32_16x16x32_f16(
      /*neg_a=*/false, a, /*neg_b=*/false, b,
      /*c_mod=*/(short)0, c, /*reuse_a=*/false, /*reuse_b=*/false);
}

static __device__ __forceinline__ float sigm(float x) {
  return 1.0f / (1.0f + __expf(-x));
}
static __device__ __forceinline__ float tanh_f(float x) {
  float e = __expf(2.0f * x);
  return (e - 1.0f) / (e + 1.0f);
}
static __device__ __forceinline__ unsigned pack2h(float a, float b) {
  union { unsigned u; _Float16 h[2]; } p;
  p.h[0] = (_Float16)a; p.h[1] = (_Float16)b;
  return p.u;
}
// Cross half-wave swap (lane <-> lane^16): ds_swizzle SWAPX16, no addr VGPR.
static __device__ __forceinline__ unsigned swap16(unsigned v) {
  return (unsigned)__builtin_amdgcn_ds_swizzle((int)v, 0x401f);
}

// One wave32 handles a 16-batch tile for the full enc+dec recurrence.
__global__ __launch_bounds__(32)
void gru_seq2seq_kernel(const float* __restrict__ x,
                        const float* __restrict__ Wemb, const float* __restrict__ bemb,
                        const float* __restrict__ WihE, const float* __restrict__ WhhE,
                        const float* __restrict__ bihE, const float* __restrict__ bhhE,
                        const float* __restrict__ WihD, const float* __restrict__ WhhD,
                        const float* __restrict__ bihD, const float* __restrict__ bhhD,
                        const float* __restrict__ Wout, const float* __restrict__ bout,
                        const int*   __restrict__ flen,
                        float* __restrict__ y) {
  const int lane = threadIdx.x & 31;
  const int half = lane >> 4;      // 0: lanes 0-15, 1: lanes 16-31
  const int ml   = lane & 15;      // batch index within tile / A-row index
  const bool low = (half == 0);
  const int b0   = blockIdx.x * 16;
  const int kb   = half * 8;       // M base (D layout)

  // h (D layout, packed pairs) -> B-operand layout; 4 ds_swizzle per call.
  // low lane holds hid {0..7}(pe) {16..23}(po); high lane {8..15}(pe) {24..31}(po).
  auto buildB = [&](const float* hDe, const float* hDo) -> v16h {
    union { v16h v; unsigned u[8]; } hb;
#pragma unroll
    for (int i = 0; i < 4; ++i) {
      unsigned pe  = pack2h(hDe[2 * i], hDe[2 * i + 1]);
      unsigned po  = pack2h(hDo[2 * i], hDo[2 * i + 1]);
      unsigned snd = low ? po : pe;      // what the partner lane needs
      unsigned rcv = swap16(snd);        // partner's pe (low) / partner's po (high)
      hb.u[i]     = low ? pe  : rcv;
      hb.u[4 + i] = low ? rcv : po;
    }
    return hb.v;
  };

  // ================= encoder-resident state =================
  // WMMA A layout, chunk g: rows = gates [16g,16g+16), K = hidden 0..31.
  v16h aIhE[6], aHhE[6];
#pragma unroll
  for (int g = 0; g < 6; ++g) {
    const float* r0 = WihE + (g * 16 + ml) * kH;
    const float* r1 = WhhE + (g * 16 + ml) * kH;
#pragma unroll
    for (int j = 0; j < 8; ++j) {
      aIhE[g][j] = (_Float16)r0[half * 16 + j];  aIhE[g][8 + j] = (_Float16)r0[half * 16 + 8 + j];
      aHhE[g][j] = (_Float16)r1[half * 16 + j];  aHhE[g][8 + j] = (_Float16)r1[half * 16 + 8 + j];
    }
  }
  // Biases as v8f D-tiles fed into the WMMA C operand (slot r -> gate g*16+r+kb).
  v8f bRZe[4], bXNe[2], bHNe[2];
#pragma unroll
  for (int g = 0; g < 4; ++g)
#pragma unroll
    for (int r = 0; r < 8; ++r)
      bRZe[g][r] = bihE[g * 16 + r + kb] + bhhE[g * 16 + r + kb];
#pragma unroll
  for (int g = 0; g < 2; ++g)
#pragma unroll
    for (int r = 0; r < 8; ++r) {
      bXNe[g][r] = bihE[(4 + g) * 16 + r + kb];
      bHNe[g][r] = bhhE[(4 + g) * 16 + r + kb];
    }
  // Embedder weights: lane computes e[batch ml][hid half*16 + i], i = 0..15.
  float we[16][4], be[16];
#pragma unroll
  for (int i = 0; i < 16; ++i) {
    int hid = half * 16 + i;
#pragma unroll
    for (int k = 0; k < 4; ++k) we[i][k] = Wemb[hid * kDin + k];
    be[i] = bemb[hid];
  }

  auto embed = [&](float4 xv) -> v16h {
    v16h e;
#pragma unroll
    for (int i = 0; i < 16; ++i) {
      float s = fmaf(we[i][0], xv.x,
                fmaf(we[i][1], xv.y,
                fmaf(we[i][2], xv.z,
                fmaf(we[i][3], xv.w, be[i]))));
      e[i] = (_Float16)fmaxf(s, 0.0f);
    }
    return e;
  };

  // hidden state in D layout: low lane holds hid {r, 16+r}, high lane {8+r, 24+r}.
  float hDe[8], hDo[8];
#pragma unroll
  for (int r = 0; r < 8; ++r) { hDe[r] = 0.0f; hDo[r] = 0.0f; }

  const float4* xp = (const float4*)x;   // [B][T] of float4 (Din==4)
  const size_t xrow = (size_t)(b0 + ml) * kT;

  // ================= encoder scan (software-pipelined x/embed) =================
  float4 xv = xp[xrow];
  v16h eB = embed(xv);
  for (int t = 0; t < kT; ++t) {
    const int tn = (t + 1 < kT) ? t + 1 : t;
    float4 xn = xp[xrow + tn];                       // issued early, used next iter
    if (t + 2 < kT) __builtin_prefetch(&xp[xrow + t + 2], 0, 1);

    v16h hB = buildB(hDe, hDo);                      // critical path: hD -> hB

    v8f c0 = wmma16(aIhE[0], eB, bRZe[0]); c0 = wmma16(aHhE[0], hB, c0);
    v8f c1 = wmma16(aIhE[1], eB, bRZe[1]); c1 = wmma16(aHhE[1], hB, c1);
    v8f c2 = wmma16(aIhE[2], eB, bRZe[2]); c2 = wmma16(aHhE[2], hB, c2);
    v8f c3 = wmma16(aIhE[3], eB, bRZe[3]); c3 = wmma16(aHhE[3], hB, c3);
    v8f x4 = wmma16(aIhE[4], eB, bXNe[0]); v8f h4 = wmma16(aHhE[4], hB, bHNe[0]);
    v8f x5 = wmma16(aIhE[5], eB, bXNe[1]); v8f h5 = wmma16(aHhE[5], hB, bHNe[1]);

    v16h eN = embed(xn);                             // co-executes with WMMAs

#pragma unroll
    for (int r = 0; r < 8; ++r) {
      float rg_e = sigm(c0[r]);
      float rg_o = sigm(c1[r]);
      float zg_e = sigm(c2[r]);
      float zg_o = sigm(c3[r]);
      float ng_e = tanh_f(fmaf(rg_e, h4[r], x4[r]));
      float ng_o = tanh_f(fmaf(rg_o, h5[r], x5[r]));
      hDe[r] = (1.0f - zg_e) * ng_e + zg_e * hDe[r];
      hDo[r] = (1.0f - zg_o) * ng_o + zg_o * hDo[r];
    }
    eB = eN;
  }

  // ================= decoder-only state (loaded after encoder loop) =================
  v16h aIhD[6], aHhD[6];
#pragma unroll
  for (int g = 0; g < 6; ++g) {
    const float* r2 = WihD + (g * 16 + ml) * kH;
    const float* r3 = WhhD + (g * 16 + ml) * kH;
#pragma unroll
    for (int j = 0; j < 8; ++j) {
      aIhD[g][j] = (_Float16)r2[half * 16 + j];  aIhD[g][8 + j] = (_Float16)r2[half * 16 + 8 + j];
      aHhD[g][j] = (_Float16)r3[half * 16 + j];  aHhD[g][8 + j] = (_Float16)r3[half * 16 + 8 + j];
    }
  }
  v8f bRZd[4], bXNd[2], bHNd[2];
#pragma unroll
  for (int g = 0; g < 4; ++g)
#pragma unroll
    for (int r = 0; r < 8; ++r)
      bRZd[g][r] = bihD[g * 16 + r + kb] + bhhD[g * 16 + r + kb];
#pragma unroll
  for (int g = 0; g < 2; ++g)
#pragma unroll
    for (int r = 0; r < 8; ++r) {
      bXNd[g][r] = bihD[(4 + g) * 16 + r + kb];
      bHNd[g][r] = bhhD[(4 + g) * 16 + r + kb];
    }
  // Output projection: lane's hidden slice is {kb..kb+7, 16+kb..16+kb+7}.
  float woE[4][8], woO[4][8], bo[4];
#pragma unroll
  for (int o = 0; o < 4; ++o) {
#pragma unroll
    for (int r = 0; r < 8; ++r) {
      woE[o][r] = Wout[o * kH + kb + r];
      woO[o][r] = Wout[o * kH + 16 + kb + r];
    }
    bo[o] = bout[o];
  }

  // ================= decoder free-run =================
  const int F = *flen;
  for (int f = 0; f < F; ++f) {
    v16h hB = buildB(hDe, hDo);

    // decoder input == hidden; both Wih_d and Whh_d consume hB
    v8f c0 = wmma16(aIhD[0], hB, bRZd[0]); c0 = wmma16(aHhD[0], hB, c0);
    v8f c1 = wmma16(aIhD[1], hB, bRZd[1]); c1 = wmma16(aHhD[1], hB, c1);
    v8f c2 = wmma16(aIhD[2], hB, bRZd[2]); c2 = wmma16(aHhD[2], hB, c2);
    v8f c3 = wmma16(aIhD[3], hB, bRZd[3]); c3 = wmma16(aHhD[3], hB, c3);
    v8f x4 = wmma16(aIhD[4], hB, bXNd[0]); v8f h4 = wmma16(aHhD[4], hB, bHNd[0]);
    v8f x5 = wmma16(aIhD[5], hB, bXNd[1]); v8f h5 = wmma16(aHhD[5], hB, bHNd[1]);

#pragma unroll
    for (int r = 0; r < 8; ++r) {
      float rg_e = sigm(c0[r]);
      float rg_o = sigm(c1[r]);
      float zg_e = sigm(c2[r]);
      float zg_o = sigm(c3[r]);
      float ng_e = tanh_f(fmaf(rg_e, h4[r], x4[r]));
      float ng_o = tanh_f(fmaf(rg_o, h5[r], x5[r]));
      hDe[r] = (1.0f - zg_e) * ng_e + zg_e * hDe[r];
      hDo[r] = (1.0f - zg_o) * ng_o + zg_o * hDo[r];
    }

    // y[b, f, 0..3] = Wout @ h + bout ; each half sums its 16 hidden, xor-16 reduce
    float yv[4];
#pragma unroll
    for (int o = 0; o < 4; ++o) {
      float p = 0.0f;
#pragma unroll
      for (int r = 0; r < 8; ++r)
        p = fmaf(woE[o][r], hDe[r], fmaf(woO[o][r], hDo[r], p));
      union { float f; unsigned u; } pu, pr;
      pu.f = p;
      pr.u = swap16(pu.u);
      yv[o] = p + pr.f + bo[o];
    }
    if (low) {
      float4* yp = (float4*)y;
      yp[(size_t)(b0 + ml) * F + f] = make_float4(yv[0], yv[1], yv[2], yv[3]);
    }
  }
}

extern "C" void kernel_launch(void* const* d_in, const int* in_sizes, int n_in,
                              void* d_out, int out_size, void* d_ws, size_t ws_size,
                              hipStream_t stream) {
  const float* x    = (const float*)d_in[0];
  const float* Wemb = (const float*)d_in[1];
  const float* bemb = (const float*)d_in[2];
  const float* WihE = (const float*)d_in[3];
  const float* WhhE = (const float*)d_in[4];
  const float* bihE = (const float*)d_in[5];
  const float* bhhE = (const float*)d_in[6];
  const float* WihD = (const float*)d_in[7];
  const float* WhhD = (const float*)d_in[8];
  const float* bihD = (const float*)d_in[9];
  const float* bhhD = (const float*)d_in[10];
  const float* Wout = (const float*)d_in[11];
  const float* bout = (const float*)d_in[12];
  const int*   flen = (const int*)d_in[13];

  const int B = in_sizes[0] / (kT * kDin);   // 2048
  dim3 grid(B / 16), block(32);
  hipLaunchKernelGGL(gru_seq2seq_kernel, grid, block, 0, stream,
                     x, Wemb, bemb, WihE, WhhE, bihE, bhhE,
                     WihD, WhhD, bihD, bhhD, Wout, bout, flen,
                     (float*)d_out);
}
